// MambaMoEBlock_4827543241060
// MI455X (gfx1250) — matrix operand order repaired and checked by reading, not compiled, not run on hardware
//
#include <hip/hip_runtime.h>

// ---------------- problem constants ----------------
#define BT      4096          // B*T tokens
#define DM      1024          // d_model
#define DIN     2048          // d_inner
#define DSTATE  128
#define NH      32            // heads
#define HD      64            // head dim
#define CONVD   2304          // d_inner + 2*d_state
#define DPROJ   4384          // 2*d_inner + 2*d_state + nheads
#define NCH     32            // chunks per batch (2048/64)
#define CHL     64            // chunk length
#define NB      2             // batch

typedef __attribute__((ext_vector_type(16))) _Float16 v16h;
typedef __attribute__((ext_vector_type(8)))  _Float16 v8h;
typedef __attribute__((ext_vector_type(8)))  float    v8f;

#define SHUF16(lo, hi) __builtin_shufflevector(lo, hi, 0,1,2,3,4,5,6,7,8,9,10,11,12,13,14,15)
#define WMMA(a, b, c) __builtin_amdgcn_wmma_f32_16x16x32_f16(false, a, false, b, (short)0, c, false, false)

// ---------------- f32 -> f16 convert (same layout) ----------------
__global__ void k_f32_to_f16(const float* __restrict__ src, _Float16* __restrict__ dst, size_t n)
{
    size_t i = (size_t)blockIdx.x * blockDim.x + threadIdx.x;
    if (i < n) dst[i] = (_Float16)src[i];
}

// ---------------- f32 [K][N] -> f16 transposed [N][K] ----------------
__global__ __launch_bounds__(256) void k_cvt_T(const float* __restrict__ src,
                                               _Float16* __restrict__ dst,
                                               int K, int N)
{
    __shared__ float tile[32][33];
    int n0 = blockIdx.x * 32, k0 = blockIdx.y * 32;
    int tid = threadIdx.x;
    int c = tid & 31, r0 = tid >> 5;
    #pragma unroll
    for (int i = 0; i < 4; ++i) {
        int r = r0 + 8*i;
        int k = k0 + r, n = n0 + c;
        tile[r][c] = (k < K && n < N) ? src[(size_t)k*N + n] : 0.f;
    }
    __syncthreads();
    #pragma unroll
    for (int i = 0; i < 4; ++i) {
        int r = r0 + 8*i;
        int n = n0 + r, k = k0 + c;
        if (n < N && k < K) dst[(size_t)n*K + k] = (_Float16)tile[c][r];
    }
}

// ---------------- WMMA GEMM, no LDS: C(f32,[M][N]) = A(f16,[M][K]) * Bt(f16,[N][K])^T ----------------
__global__ __launch_bounds__(256) void k_gemm_f16t(const _Float16* __restrict__ A,
                                                   const _Float16* __restrict__ Bt,
                                                   float* __restrict__ C,
                                                   int M, int N, int K)
{
    const int lane = threadIdx.x & 31;
    const int wid  = threadIdx.x >> 5;
    const int wm   = wid & 1;
    const int wn   = wid >> 1;
    const int m0   = blockIdx.y * 128 + wm * 64;
    const int n0   = blockIdx.x * 128 + wn * 32;

    const int rl   = lane & 15;
    const int hi16 = lane >> 4;
    const int kbA  = hi16 * 16;
    const int kbB  = hi16 * 16;

    const _Float16* Ap[4];
    #pragma unroll
    for (int f = 0; f < 4; ++f) Ap[f] = A + (size_t)(m0 + f*16 + rl) * K;

    int nb0 = n0 + rl;       if (nb0 >= N) nb0 = 0;
    int nb1 = n0 + 16 + rl;  if (nb1 >= N) nb1 = 0;
    const _Float16* Bp0 = Bt + (size_t)nb0 * K;
    const _Float16* Bp1 = Bt + (size_t)nb1 * K;

    v8f acc[4][2];
    #pragma unroll
    for (int f = 0; f < 4; ++f) { v8f z = {}; acc[f][0] = z; acc[f][1] = z; }

    for (int k0 = 0; k0 < K; k0 += 32) {
        v8h bl0 = *(const v8h*)(Bp0 + k0 + kbB);
        v8h bh0 = *(const v8h*)(Bp0 + k0 + kbB + 8);
        v8h bl1 = *(const v8h*)(Bp1 + k0 + kbB);
        v8h bh1 = *(const v8h*)(Bp1 + k0 + kbB + 8);
        v16h bf0 = SHUF16(bl0, bh0);
        v16h bf1 = SHUF16(bl1, bh1);
        #pragma unroll
        for (int f = 0; f < 4; ++f) {
            v8h al = *(const v8h*)(Ap[f] + k0 + kbA);
            v8h ah = *(const v8h*)(Ap[f] + k0 + kbA + 8);
            v16h af = SHUF16(al, ah);
            acc[f][0] = WMMA(af, bf0, acc[f][0]);
            acc[f][1] = WMMA(af, bf1, acc[f][1]);
        }
    }

    const int nc0 = n0 + rl;
    const int nc1 = n0 + 16 + rl;
    #pragma unroll
    for (int f = 0; f < 4; ++f) {
        int mb = m0 + f*16 + 8*hi16;
        #pragma unroll
        for (int v = 0; v < 8; ++v) {
            int m = mb + v;
            if (nc0 < N) C[(size_t)m * N + nc0] = acc[f][0][v];
            if (nc1 < N) C[(size_t)m * N + nc1] = acc[f][1][v];
        }
    }
}

// ---------------- router softmax: one wave per token ----------------
__global__ __launch_bounds__(256) void k_router(const float* __restrict__ x,
                                                const float* __restrict__ rw,
                                                float* __restrict__ routing)
{
    int wid = threadIdx.x >> 5, lane = threadIdx.x & 31;
    int tok = blockIdx.x * 8 + wid;
    if (tok >= BT) return;
    float a0 = 0.f, a1 = 0.f, a2 = 0.f, a3 = 0.f;
    const float* xr = x + (size_t)tok * DM;
    for (int d = lane; d < DM; d += 32) {
        float xv = xr[d];
        a0 += xv * rw[d*4+0]; a1 += xv * rw[d*4+1];
        a2 += xv * rw[d*4+2]; a3 += xv * rw[d*4+3];
    }
    #pragma unroll
    for (int off = 16; off > 0; off >>= 1) {
        a0 += __shfl_down(a0, off, 32); a1 += __shfl_down(a1, off, 32);
        a2 += __shfl_down(a2, off, 32); a3 += __shfl_down(a3, off, 32);
    }
    if (lane == 0) {
        float m = fmaxf(fmaxf(a0, a1), fmaxf(a2, a3));
        float e0 = expf(a0-m), e1 = expf(a1-m), e2 = expf(a2-m), e3 = expf(a3-m);
        float s = e0 + e1 + e2 + e3;
        routing[tok*4+0] = e0/s; routing[tok*4+1] = e1/s;
        routing[tok*4+2] = e2/s; routing[tok*4+3] = e3/s;
    }
}

// ---------------- depthwise conv + SiLU + split (f16 B/C) + dt/a ----------------
__global__ __launch_bounds__(256) void k_conv_split(const float* __restrict__ zx,
    const float* __restrict__ cw, const float* __restrict__ cb,
    const float* __restrict__ dtb, const float* __restrict__ alog,
    float* __restrict__ xs, float* __restrict__ dts, float* __restrict__ as_,
    _Float16* __restrict__ Bm16, _Float16* __restrict__ Cm16)
{
    int tok = blockIdx.x;
    int b = tok >> 11, t = tok & 2047;
    int tid = threadIdx.x;
    if (tid < NH) {
        float raw = zx[(size_t)tok * DPROJ + (DPROJ - NH) + tid] + dtb[tid];
        float dt  = (raw > 20.f) ? raw : log1pf(expf(raw));   // softplus
        dts[tok*NH + tid] = dt;
        as_[tok*NH + tid] = dt * (-expf(alog[tid]));          // dt * A
    }
    for (int ch = tid; ch < CONVD; ch += 256) {
        float acc = cb[ch];
        #pragma unroll
        for (int k = 0; k < 4; ++k) {
            int tt = t - 3 + k;
            if (tt >= 0)
                acc += cw[ch*4 + k] * zx[(size_t)(b*2048 + tt) * DPROJ + DIN + ch];
        }
        float v = acc / (1.f + expf(-acc));                   // silu
        if (ch < DIN)                xs[(size_t)tok * DIN + ch] = v;
        else if (ch < DIN + DSTATE)  Bm16[(size_t)tok*DSTATE + (ch - DIN)] = (_Float16)v;
        else                         Cm16[(size_t)tok*DSTATE + (ch - DIN - DSTATE)] = (_Float16)v;
    }
}

// ---------------- per-chunk cumsum of a = dt*A; chunk decay ----------------
__global__ void k_cumsum(const float* __restrict__ a, float* __restrict__ acs,
                         float* __restrict__ cd)
{
    int g = blockIdx.x * blockDim.x + threadIdx.x;
    if (g >= NB*NCH*NH) return;
    int h = g & 31, bc = g >> 5;
    int b = bc >> 5, c = bc & 31;
    int tb = b*2048 + c*64;
    float run = 0.f;
    for (int l = 0; l < CHL; ++l) {
        run += a[(tb + l)*NH + h];
        acs[(tb + l)*NH + h] = run;
    }
    cd[bc*NH + h] = expf(run);
}

// ---------------- WMMA chunk states: states[p,n] = sum_l xdt[l,p]*wdec[l]*B[l,n] ----------------
// TN GEMM 64x128, K=64; xdtT and wdec-folded B^T staged in LDS, 8 WMMA per wave.
__global__ __launch_bounds__(256) void k_states_wmma(const float* __restrict__ xs,
    const float* __restrict__ dts, const _Float16* __restrict__ Bm16,
    const float* __restrict__ acs, float* __restrict__ states)
{
    __shared__ _Float16 xdtT[64][72];   // [p][l]
    __shared__ _Float16 BwT[128][72];   // [n][l]
    __shared__ float wdec[64];
    int blk = blockIdx.x;               // (b*32+c)*32+h
    int h = blk & 31, c = (blk >> 5) & 31, b = blk >> 10;
    int tb = b*2048 + c*64;
    int tid = threadIdx.x;
    float alast = acs[(tb + 63)*NH + h];
    if (tid < 64) wdec[tid] = expf(alast - acs[(tb + tid)*NH + h]);
    __syncthreads();
    for (int i = 0; i < 16; ++i) {
        int idx = tid + 256*i;
        int l = idx >> 6, p = idx & 63;
        xdtT[p][l] = (_Float16)(xs[(size_t)(tb + l)*DIN + h*HD + p] * dts[(tb + l)*NH + h]);
    }
    for (int i = 0; i < 32; ++i) {
        int idx = tid + 256*i;
        int l = idx >> 7, n = idx & 127;
        BwT[n][l] = (_Float16)((float)Bm16[(size_t)(tb + l)*DSTATE + n] * wdec[l]);
    }
    __syncthreads();
    const int lane = tid & 31, wid = tid >> 5;
    const int pt = wid & 3, ng = wid >> 2;
    const int rl = lane & 15, hi16 = lane >> 4;
    const int kbA = hi16*16, kbB = hi16*16;
    v8f acc[4];
    #pragma unroll
    for (int t = 0; t < 4; ++t) { v8f z = {}; acc[t] = z; }
    #pragma unroll
    for (int k0 = 0; k0 < 64; k0 += 32) {
        v8h al = *(const v8h*)&xdtT[pt*16 + rl][k0 + kbA];
        v8h ah = *(const v8h*)&xdtT[pt*16 + rl][k0 + kbA + 8];
        v16h af = SHUF16(al, ah);
        #pragma unroll
        for (int t = 0; t < 4; ++t) {
            int n = ng*64 + t*16 + rl;
            v8h bl = *(const v8h*)&BwT[n][k0 + kbB];
            v8h bh = *(const v8h*)&BwT[n][k0 + kbB + 8];
            v16h bf = SHUF16(bl, bh);
            acc[t] = WMMA(af, bf, acc[t]);
        }
    }
    size_t base = (size_t)blk * (HD*DSTATE);
    const int pb = pt*16 + 8*hi16;
    #pragma unroll
    for (int t = 0; t < 4; ++t) {
        int n = ng*64 + t*16 + rl;
        #pragma unroll
        for (int v = 0; v < 8; ++v)
            states[base + (size_t)(pb + v)*DSTATE + n] = acc[t][v];
    }
}

// ---------------- inter-chunk scan: read f32 states, emit f16 prev (carry-in) ----------------
__global__ __launch_bounds__(256) void k_scan(const float* __restrict__ states,
                                              _Float16* __restrict__ prev16,
                                              const float* __restrict__ cd)
{
    int b = blockIdx.x >> 5, h = blockIdx.x & 31;
    int tid = threadIdx.x;
    float carry[32];
    #pragma unroll
    for (int i = 0; i < 32; ++i) carry[i] = 0.f;
    for (int c = 0; c < NCH; ++c) {
        float dec = cd[(b*NCH + c)*NH + h];
        size_t base = (size_t)((b*NCH + c)*NH + h) * (HD*DSTATE);
        #pragma unroll
        for (int i = 0; i < 32; ++i) {
            size_t off = base + tid + 256*i;
            float st = states[off];
            prev16[off] = (_Float16)carry[i];
            carry[i] = st + dec * carry[i];
        }
    }
}

// ---------------- WMMA Y: Yd = (L o C.B^T) @ xdt ; Yoff = exp(acs)*C@prev^T ; +D*xs ----------------
__global__ __launch_bounds__(256) void k_y_wmma(const float* __restrict__ xs,
    const float* __restrict__ dts, const _Float16* __restrict__ Bm16,
    const _Float16* __restrict__ Cm16, const float* __restrict__ acs,
    const _Float16* __restrict__ prev16, const float* __restrict__ Dsk,
    float* __restrict__ y)
{
    __shared__ _Float16 G16[64][72];    // masked decay-weighted C.B^T
    __shared__ _Float16 xdtT[64][72];   // [p][l]
    __shared__ float acsL[64];
    __shared__ float exL[64];
    int blk = blockIdx.x;
    int h = blk & 31, c = (blk >> 5) & 31, b = blk >> 10;
    int tb = b*2048 + c*64;
    int tid = threadIdx.x;
    if (tid < 64) {
        float av = acs[(tb + tid)*NH + h];
        acsL[tid] = av; exL[tid] = expf(av);
    }
    for (int i = 0; i < 16; ++i) {
        int idx = tid + 256*i;
        int l = idx >> 6, p = idx & 63;
        xdtT[p][l] = (_Float16)(xs[(size_t)(tb + l)*DIN + h*HD + p] * dts[(tb + l)*NH + h]);
    }
    __syncthreads();

    const int lane = tid & 31, wid = tid >> 5;
    const int rl = lane & 15, hi16 = lane >> 4;
    const int kbA = hi16*16, kbB = hi16*16;

    // phase 1: G = C @ B^T (64x64, K=128) from global, mask+decay in epilogue -> G16
    {
        const int lt = wid & 3, sg = wid >> 2;
        const _Float16* Ap  = Cm16 + (size_t)(tb + lt*16 + rl)*DSTATE;
        const _Float16* Bp0 = Bm16 + (size_t)(tb + sg*32 + rl)*DSTATE;
        const _Float16* Bp1 = Bm16 + (size_t)(tb + sg*32 + 16 + rl)*DSTATE;
        v8f g0 = {}, g1 = {};
        #pragma unroll
        for (int k0 = 0; k0 < 128; k0 += 32) {
            v8h al = *(const v8h*)(Ap + k0 + kbA);
            v8h ah = *(const v8h*)(Ap + k0 + kbA + 8);
            v16h af = SHUF16(al, ah);
            v8h bl0 = *(const v8h*)(Bp0 + k0 + kbB);
            v8h bh0 = *(const v8h*)(Bp0 + k0 + kbB + 8);
            v16h bf0 = SHUF16(bl0, bh0);
            v8h bl1 = *(const v8h*)(Bp1 + k0 + kbB);
            v8h bh1 = *(const v8h*)(Bp1 + k0 + kbB + 8);
            v16h bf1 = SHUF16(bl1, bh1);
            g0 = WMMA(af, bf0, g0);
            g1 = WMMA(af, bf1, g1);
        }
        #pragma unroll
        for (int v = 0; v < 8; ++v) {
            int l  = lt*16 + v + 8*hi16;
            int s0 = sg*32 + rl, s1 = sg*32 + 16 + rl;
            float e0 = (s0 <= l) ? expf(acsL[l] - acsL[s0]) * g0[v] : 0.f;
            float e1 = (s1 <= l) ? expf(acsL[l] - acsL[s1]) * g1[v] : 0.f;
            G16[l][s0] = (_Float16)e0;
            G16[l][s1] = (_Float16)e1;
        }
    }
    __syncthreads();

    // phase 2: Yd(K=64, LDS) + Yoff(K=128, global prev16)
    {
        const int lt = wid & 3, pg = wid >> 2;
        const int l0 = lt*16 + rl;
        const int p0 = pg*32 + rl, p1 = pg*32 + 16 + rl;
        const _Float16* Ap  = Cm16 + (size_t)(tb + l0)*DSTATE;
        size_t pbase = (size_t)blk * (HD*DSTATE);
        const _Float16* Pp0 = prev16 + pbase + (size_t)p0*DSTATE;
        const _Float16* Pp1 = prev16 + pbase + (size_t)p1*DSTATE;
        v8f d0 = {}, d1 = {}, o0 = {}, o1 = {};
        #pragma unroll
        for (int k0 = 0; k0 < 64; k0 += 32) {
            v8h al = *(const v8h*)&G16[l0][k0 + kbA];
            v8h ah = *(const v8h*)&G16[l0][k0 + kbA + 8];
            v16h af = SHUF16(al, ah);
            v8h bl0 = *(const v8h*)&xdtT[p0][k0 + kbB];
            v8h bh0 = *(const v8h*)&xdtT[p0][k0 + kbB + 8];
            v16h bf0 = SHUF16(bl0, bh0);
            v8h bl1 = *(const v8h*)&xdtT[p1][k0 + kbB];
            v8h bh1 = *(const v8h*)&xdtT[p1][k0 + kbB + 8];
            v16h bf1 = SHUF16(bl1, bh1);
            d0 = WMMA(af, bf0, d0);
            d1 = WMMA(af, bf1, d1);
        }
        #pragma unroll
        for (int k0 = 0; k0 < 128; k0 += 32) {
            v8h al = *(const v8h*)(Ap + k0 + kbA);
            v8h ah = *(const v8h*)(Ap + k0 + kbA + 8);
            v16h af = SHUF16(al, ah);
            v8h bl0 = *(const v8h*)(Pp0 + k0 + kbB);
            v8h bh0 = *(const v8h*)(Pp0 + k0 + kbB + 8);
            v16h bf0 = SHUF16(bl0, bh0);
            v8h bl1 = *(const v8h*)(Pp1 + k0 + kbB);
            v8h bh1 = *(const v8h*)(Pp1 + k0 + kbB + 8);
            v16h bf1 = SHUF16(bl1, bh1);
            o0 = WMMA(af, bf0, o0);
            o1 = WMMA(af, bf1, o1);
        }
        float dsk = Dsk[h];
        #pragma unroll
        for (int v = 0; v < 8; ++v) {
            int l = lt*16 + v + 8*hi16;
            float ex = exL[l];
            size_t row = (size_t)(tb + l)*DIN + h*HD;
            y[row + p0] = d0[v] + ex*o0[v] + dsk * xs[row + p0];
            y[row + p1] = d1[v] + ex*o1[v] + dsk * xs[row + p1];
        }
    }
}

// ---------------- gate (silu(z)) + RMSNorm -> f16 ----------------
__global__ __launch_bounds__(256) void k_gate(const float* __restrict__ zx,
    const float* __restrict__ y, const float* __restrict__ nw, _Float16* __restrict__ y16)
{
    __shared__ float red[256];
    int tok = blockIdx.x, tid = threadIdx.x;
    float yv[8];
    float ss = 0.f;
    #pragma unroll
    for (int i = 0; i < 8; ++i) {
        int d = tid + 256*i;
        float z = zx[(size_t)tok*DPROJ + d];
        float g = z / (1.f + expf(-z));
        float v = y[(size_t)tok*DIN + d] * g;
        yv[i] = v;
        ss += v*v;
    }
    red[tid] = ss;
    __syncthreads();
    for (int off = 128; off > 0; off >>= 1) {
        if (tid < off) red[tid] += red[tid + off];
        __syncthreads();
    }
    float scale = rsqrtf(red[0] / (float)DIN + 1e-5f);
    #pragma unroll
    for (int i = 0; i < 8; ++i) {
        int d = tid + 256*i;
        y16[(size_t)tok*DIN + d] = (_Float16)(yv[i] * scale * nw[d]);
    }
}

// ---------------- router-weighted combine ----------------
__global__ void k_combine(const float* __restrict__ oute, const float* __restrict__ routing,
                          float* __restrict__ out, int e)
{
    size_t idx = (size_t)blockIdx.x * blockDim.x + threadIdx.x;
    if (idx >= (size_t)BT * DM) return;
    int tok = (int)(idx >> 10);
    float v = routing[tok*4 + e] * oute[idx];
    if (e == 0) out[idx] = v; else out[idx] += v;
}

// ---------------- host launcher ----------------
extern "C" void kernel_launch(void* const* d_in, const int* in_sizes, int n_in,
                              void* d_out, int out_size, void* d_ws, size_t ws_size,
                              hipStream_t stream)
{
    const float* x        = (const float*)d_in[0];
    const float* router_w = (const float*)d_in[1];
    const float* in_w     = (const float*)d_in[2];
    const float* conv_w   = (const float*)d_in[3];
    const float* conv_b   = (const float*)d_in[4];
    const float* dt_bias  = (const float*)d_in[5];
    const float* A_log    = (const float*)d_in[6];
    const float* D_skip   = (const float*)d_in[7];
    const float* norm_w   = (const float*)d_in[8];
    const float* out_w    = (const float*)d_in[9];
    float* out = (float*)d_out;

    char* w = (char*)d_ws;
    size_t off = 0;
    auto alloc = [&](size_t bytes) -> char* {
        char* p = w + off;
        off += (bytes + 255) & ~(size_t)255;
        return p;
    };
    _Float16* x16     = (_Float16*)alloc((size_t)BT*DM*2);
    _Float16* inw16t  = (_Float16*)alloc((size_t)DM*DPROJ*2);   // [DPROJ][DM]
    _Float16* outw16t = (_Float16*)alloc((size_t)DIN*DM*2);     // [DM][DIN]
    float* zx      = (float*)alloc((size_t)BT*DPROJ*4);
    float* xsb     = (float*)alloc((size_t)BT*DIN*4);
    _Float16* bm16 = (_Float16*)alloc((size_t)BT*DSTATE*2);
    _Float16* cm16 = (_Float16*)alloc((size_t)BT*DSTATE*2);
    float* dtb_    = (float*)alloc((size_t)BT*NH*4);
    float* ab      = (float*)alloc((size_t)BT*NH*4);
    float* acsb    = (float*)alloc((size_t)BT*NH*4);
    float* cdb     = (float*)alloc((size_t)NB*NCH*NH*4);
    float* statesb = (float*)alloc((size_t)NB*NCH*NH*HD*DSTATE*4);
    _Float16* prev16 = (_Float16*)alloc((size_t)NB*NCH*NH*HD*DSTATE*2);
    float* yb      = (float*)alloc((size_t)BT*DIN*4);
    _Float16* y16b = (_Float16*)alloc((size_t)BT*DIN*2);
    float* outeb   = (float*)alloc((size_t)BT*DM*4);
    float* routing = (float*)alloc((size_t)BT*4*4);

    { size_t n = (size_t)BT*DM;
      k_f32_to_f16<<<(unsigned)((n+255)/256), 256, 0, stream>>>(x, x16, n); }
    k_router<<<BT/8, 256, 0, stream>>>(x, router_w, routing);

    for (int e = 0; e < 4; ++e) {
        { dim3 g((DPROJ+31)/32, (DM+31)/32);
          k_cvt_T<<<g, 256, 0, stream>>>(in_w + (size_t)e*DM*DPROJ, inw16t, DM, DPROJ); }
        { dim3 g1((DPROJ + 127)/128, BT/128);
          k_gemm_f16t<<<g1, 256, 0, stream>>>(x16, inw16t, zx, BT, DPROJ, DM); }

        k_conv_split<<<BT, 256, 0, stream>>>(zx,
            conv_w + (size_t)e*CONVD*4, conv_b + (size_t)e*CONVD,
            dt_bias + (size_t)e*NH, A_log + (size_t)e*NH,
            xsb, dtb_, ab, bm16, cm16);

        k_cumsum<<<(NB*NCH*NH + 255)/256, 256, 0, stream>>>(ab, acsb, cdb);
        k_states_wmma<<<NB*NCH*NH, 256, 0, stream>>>(xsb, dtb_, bm16, acsb, statesb);
        k_scan<<<NB*NH, 256, 0, stream>>>(statesb, prev16, cdb);
        k_y_wmma<<<NB*NCH*NH, 256, 0, stream>>>(xsb, dtb_, bm16, cm16, acsb, prev16,
                                                D_skip + (size_t)e*NH, yb);
        k_gate<<<BT, 256, 0, stream>>>(zx, yb, norm_w + (size_t)e*DIN, y16b);

        { dim3 g((DM+31)/32, (DIN+31)/32);
          k_cvt_T<<<g, 256, 0, stream>>>(out_w + (size_t)e*DIN*DM, outw16t, DIN, DM); }
        { dim3 g2(DM/128, BT/128);
          k_gemm_f16t<<<g2, 256, 0, stream>>>(y16b, outw16t, outeb, BT, DM, DIN); }

        { size_t n = (size_t)BT*DM;
          k_combine<<<(unsigned)((n+255)/256), 256, 0, stream>>>(outeb, routing, out, e); }
    }
}
